// OfficialVMambaVSSBlock_46248207843794
// MI455X (gfx1250) — matrix operand order repaired and checked by reading, not compiled, not run on hardware
//
#include <hip/hip_runtime.h>
#include <math.h>

// ---------------- problem constants ----------------
#define B_   8
#define H_   56
#define W_   56
#define C_   96
#define DIN  192
#define KDIR 4
#define NST  16
#define RNK  6
#define L_   (H_ * W_)          // 3136
#define M_   (B_ * L_)          // 25088 rows (multiple of 128)

typedef __bf16 bf16;
typedef __attribute__((ext_vector_type(16))) __bf16 v16bf;
typedef __attribute__((ext_vector_type(8)))  float  v8f;

// ---------------- small helpers ----------------
__device__ __forceinline__ float siluf(float x) {
    return x / (1.0f + __expf(-x));
}
__device__ __forceinline__ float softplusf(float x) {
    return (x > 20.0f) ? x : log1pf(__expf(x));
}
__device__ __forceinline__ float gelu_exact(float x) {
    return 0.5f * x * (1.0f + erff(x * 0.70710678118654752f));
}

// ============================================================================
// Weight repack: row-major f32 (K x N) -> bf16 WMMA B-fragment order:
//   frag[nt][kt][lane][j], nt = n>>4, kt = k>>5,
//   lane = (n&15) + 16*((k>>4)&1), j = k&15
// so a wave's 16x32 B fragment is 32 lanes x 16 contiguous bf16 (32B/lane).
// ============================================================================
__global__ void repack_wB_kernel(const float* __restrict__ src, bf16* __restrict__ dst,
                                 int K, int N) {
    int i = blockIdx.x * blockDim.x + threadIdx.x;
    if (i >= K * N) return;
    int k = i / N, n = i % N;
    int idx = ((((n >> 4) * (K >> 5) + (k >> 5)) * 32) +
               ((n & 15) + (((k >> 4) & 1) << 4))) * 16 + (k & 15);
    dst[idx] = (bf16)src[i];
}

// x_proj_w (KDIR, 38, DIN) -> per-direction fragment-ordered B (K=DIN x N=64, zero pad)
__global__ void repack_xproj_frag_kernel(const float* __restrict__ src, bf16* __restrict__ dst) {
    int i = blockIdx.x * blockDim.x + threadIdx.x;
    int total = KDIR * DIN * 64;
    if (i >= total) return;
    int c  = i & 63;              // n (output channel, 38 valid)
    int d  = (i >> 6) % DIN;      // k (reduction dim)
    int k4 = i / (64 * DIN);
    float v = (c < RNK + 2 * NST) ? src[(k4 * (RNK + 2 * NST) + c) * DIN + d] : 0.0f;
    int idx = ((((c >> 4) * (DIN >> 5) + (d >> 5)) * 32) +
               ((c & 15) + (((d >> 4) & 1) << 4))) * 16 + (d & 15);
    dst[(size_t)k4 * (DIN * 64) + idx] = (bf16)v;
}

__global__ void zero_f32_kernel(float* __restrict__ p, int n) {
    int i = blockIdx.x * blockDim.x + threadIdx.x;
    int stride = gridDim.x * blockDim.x;
    for (; i < n; i += stride) p[i] = 0.0f;
}

// ---------------- layernorm rows -> bf16 (one wave per row) ----------------
__global__ void ln_rows_kernel(const float* __restrict__ X, const float* __restrict__ w,
                               const float* __restrict__ b, bf16* __restrict__ Y,
                               int rows, int C) {
    int lane = threadIdx.x & 31;
    int wv   = threadIdx.x >> 5;
    int row  = blockIdx.x * (blockDim.x >> 5) + wv;
    if (row >= rows) return;
    const float* xr = X + (size_t)row * C;
    float s = 0.f, sq = 0.f;
    for (int c = lane; c < C; c += 32) { float v = xr[c]; s += v; sq += v * v; }
    #pragma unroll
    for (int off = 16; off; off >>= 1) { s += __shfl_xor(s, off); sq += __shfl_xor(sq, off); }
    float mean = s / (float)C;
    float var  = sq / (float)C - mean * mean;
    float rstd = rsqrtf(var + 1e-5f);
    bf16* yr = Y + (size_t)row * C;
    for (int c = lane; c < C; c += 32)
        yr[c] = (bf16)((xr[c] - mean) * rstd * w[c] + b[c]);
}

// ---------------- out-norm LN + *silu(z) -> bf16 ----------------
__global__ void outnorm_silu_kernel(const float* __restrict__ Yc, const float* __restrict__ w,
                                    const float* __restrict__ b, const float* __restrict__ XZ,
                                    bf16* __restrict__ Y) {
    int lane = threadIdx.x & 31;
    int wv   = threadIdx.x >> 5;
    int row  = blockIdx.x * (blockDim.x >> 5) + wv;
    if (row >= M_) return;
    const float* xr = Yc + (size_t)row * DIN;
    float s = 0.f, sq = 0.f;
    for (int c = lane; c < DIN; c += 32) { float v = xr[c]; s += v; sq += v * v; }
    #pragma unroll
    for (int off = 16; off; off >>= 1) { s += __shfl_xor(s, off); sq += __shfl_xor(sq, off); }
    float mean = s / (float)DIN;
    float var  = sq / (float)DIN - mean * mean;
    float rstd = rsqrtf(var + 1e-5f);
    const float* zr = XZ + (size_t)row * (2 * DIN) + DIN;
    bf16* yr = Y + (size_t)row * DIN;
    for (int c = lane; c < DIN; c += 32) {
        float ln = (xr[c] - mean) * rstd * w[c] + b[c];
        yr[c] = (bf16)(ln * siluf(zr[c]));
    }
}

// ---------------- depthwise 3x3 conv + bias + silu ----------------
__global__ void dwconv_silu_kernel(const float* __restrict__ XZ, const float* __restrict__ cw,
                                   const float* __restrict__ cb,
                                   float* __restrict__ XCF, bf16* __restrict__ XCB) {
    int idx = blockIdx.x * blockDim.x + threadIdx.x;
    if (idx >= B_ * L_ * DIN) return;
    int d  = idx % DIN;
    int t  = idx / DIN;
    int wx = t % W_;
    int hy = (t / W_) % H_;
    int b  = t / L_;
    float acc = cb[d];
    #pragma unroll
    for (int dh = -1; dh <= 1; ++dh) {
        int h2 = hy + dh;
        if ((unsigned)h2 >= H_) continue;
        #pragma unroll
        for (int dw = -1; dw <= 1; ++dw) {
            int w2 = wx + dw;
            if ((unsigned)w2 >= W_) continue;
            float v = XZ[((size_t)(b * L_ + h2 * W_ + w2)) * (2 * DIN) + d];
            acc += v * cw[d * 9 + (dh + 1) * 3 + (dw + 1)];
        }
    }
    float s = siluf(acc);
    size_t o = (size_t)(b * L_ + hy * W_ + wx) * DIN + d;
    XCF[o] = s;
    XCB[o] = (bf16)s;
}

// ============================================================================
// WMMA bf16 GEMM: C(MxN) = A(MxK, row-major bf16) * Bfrag(fragment-ordered bf16)
//
// Block = 256 threads = 8 waves, tile 128(M) x 32(N), BK = 32.
// Each wave owns TWO m-subtiles (wm, wm+4) sharing one register-resident B
// fragment -> 2 WMMAs per K-step per wave.
//
// A tiles are staged into a DOUBLE-BUFFERED LDS region in fragment order using
// GLOBAL_LOAD_ASYNC_TO_LDS_B128 (2 copies/thread/K-step). The steady-state
// loop is branch-free: it unconditionally issues the copy for tile kt+1,
// computes tile kt, then waits ASYNCcnt and barriers. The last tile is peeled
// (no issue / no wait / no trailing barrier). Requires ktiles >= 2 (true for
// every GEMM in this block: K = 96/192/384).
// B fragments (pre-swizzled weights) come straight from global
// (2x global_load_b128) with global_prefetch on the next K-step.
//
// MODE 0: Cf = acc
// MODE 1: Cf = acc + resid
// MODE 2: Cb = gelu(acc + bias[n])   (bf16 out)
// MODE 3: Cf = acc + bias[n] + resid
// ============================================================================
template <int MODE>
__global__ void gemm_bf16_kernel(const bf16* __restrict__ A, const bf16* __restrict__ Bf,
                                 float* __restrict__ Cf, bf16* __restrict__ Cb,
                                 const float* __restrict__ bias, const float* __restrict__ resid,
                                 int M, int N, int K, int ldc) {
    __shared__ __align__(32) bf16 Asf[2][8][32][16];   // double-buffered 128x32 A tile

    const int tid  = threadIdx.x;
    const int lane = tid & 31;
    const int wv   = tid >> 5;            // 8 waves
    const int wm   = wv & 3;              // wave computes m-subtiles wm and wm+4
    const int wn   = wv >> 2;             // wave n-subtile (16 cols)
    const int m0   = blockIdx.x * 128;
    const int ntile  = blockIdx.y * 2 + wn;   // global 16-column tile index
    const int ktiles = K >> 5;

    // staging mapping: this thread performs fragment-chunk copies i and i+256
    const int i0  = tid,        i1  = tid + 256;
    const int mt0 = i0 >> 6,    mt1 = i1 >> 6;
    const int ln0 = (i0 >> 1) & 31, ln1 = (i1 >> 1) & 31;
    const int hf0 = i0 & 1,     hf1 = i1 & 1;
    const size_t srow0 = (size_t)(m0 + mt0 * 16 + (ln0 & 15));
    const size_t srow1 = (size_t)(m0 + mt1 * 16 + (ln1 & 15));
    const int skof0 = ((ln0 >> 4) * 16) + hf0 * 8;
    const int skof1 = ((ln1 >> 4) * 16) + hf1 * 8;
    const unsigned la0 = (unsigned)(unsigned long long)&Asf[0][mt0][ln0][hf0 * 8];
    const unsigned la1 = (unsigned)(unsigned long long)&Asf[0][mt1][ln1][hf1 * 8];
    const unsigned bufbytes = 8u * 32u * 16u * 2u;     // 8 KB per buffer

    auto issue_copy = [&](int buf, int kt) {
        unsigned long long ga0 =
            (unsigned long long)(const void*)(A + srow0 * K + ((size_t)kt * 32 + skof0));
        unsigned long long ga1 =
            (unsigned long long)(const void*)(A + srow1 * K + ((size_t)kt * 32 + skof1));
        unsigned d0 = la0 + (unsigned)buf * bufbytes;
        unsigned d1 = la1 + (unsigned)buf * bufbytes;
        asm volatile("global_load_async_to_lds_b128 %0, %1, off"
                     :: "v"(d0), "v"(ga0) : "memory");
        asm volatile("global_load_async_to_lds_b128 %0, %1, off"
                     :: "v"(d1), "v"(ga1) : "memory");
    };

    auto b_frag_ptr = [&](int kt) {
        return (const v16bf*)(Bf + (((size_t)ntile * ktiles + kt) * 32 + lane) * 16);
    };

    v8f acc0 = {}, acc1 = {};

    issue_copy(0, 0);
    asm volatile("s_wait_asynccnt 0x0" ::: "memory");
    __syncthreads();

    // steady state: branch-free body, last tile peeled (ktiles >= 2 always)
    int kt = 0;
    for (; kt < ktiles - 1; ++kt) {
        const int cur = kt & 1;
        issue_copy(cur ^ 1, kt + 1);                         // overlap copy with compute

        v16bf bfr = *b_frag_ptr(kt);
        __builtin_prefetch(b_frag_ptr(kt + 1), 0, 1);        // global_prefetch next B

        v16bf af0 = *(const v16bf*)(&Asf[cur][wm][lane][0]);       // 2x ds_load_b128
        v16bf af1 = *(const v16bf*)(&Asf[cur][wm + 4][lane][0]);   // 2x ds_load_b128

        acc0 = __builtin_amdgcn_wmma_f32_16x16x32_bf16(false, af0, false, bfr,
                                                       (short)0, acc0, false, false);
        acc1 = __builtin_amdgcn_wmma_f32_16x16x32_bf16(false, af1, false, bfr,
                                                       (short)0, acc1, false, false);

        asm volatile("s_wait_asynccnt 0x0" ::: "memory");
        __syncthreads();
    }

    // final K-tile: no issue, no wait, no trailing barrier
    {
        const int cur = kt & 1;
        v16bf bfr = *b_frag_ptr(kt);
        v16bf af0 = *(const v16bf*)(&Asf[cur][wm][lane][0]);
        v16bf af1 = *(const v16bf*)(&Asf[cur][wm + 4][lane][0]);
        acc0 = __builtin_amdgcn_wmma_f32_16x16x32_bf16(false, af0, false, bfr,
                                                       (short)0, acc0, false, false);
        acc1 = __builtin_amdgcn_wmma_f32_16x16x32_bf16(false, af1, false, bfr,
                                                       (short)0, acc1, false, false);
    }

    // C/D layout: VGPR i holds row (lane>=16 ? 8 : 0)+i, col = lane&15
    const int gn = ntile * 16 + (lane & 15);
    #pragma unroll
    for (int half = 0; half < 2; ++half) {
        v8f acc = (half == 0) ? acc0 : acc1;
        int gmbase = m0 + (wm + half * 4) * 16 + (lane >> 4) * 8;
        #pragma unroll
        for (int i = 0; i < 8; ++i) {
            int gm = gmbase + i;
            float v = acc[i];
            size_t o = (size_t)gm * ldc + gn;
            if (MODE == 0) {
                Cf[o] = v;
            } else if (MODE == 1) {
                Cf[o] = v + resid[o];
            } else if (MODE == 2) {
                Cb[o] = (bf16)gelu_exact(v + bias[gn]);
            } else { // MODE 3
                Cf[o] = v + bias[gn] + resid[o];
            }
        }
    }
}

// ---------------- selective scan (sequential in L, parallel over b,k,d) ------
// One block per (b,k), 192 threads = DIN channels. Each thread carries h[16].
__global__ void scan_kernel(const float* __restrict__ proj,    // 4 x M x 64
                            const float* __restrict__ XCF,     // M x DIN
                            const float* __restrict__ dt_w,    // K x DIN x R
                            const float* __restrict__ dt_b,    // K x DIN
                            const float* __restrict__ A_logs,  // K x DIN x N
                            const float* __restrict__ Ds,      // K x DIN
                            float* __restrict__ Yc) {          // M x DIN (atomic acc)
    const int d = threadIdx.x;
    const int k = blockIdx.x % KDIR;
    const int b = blockIdx.x / KDIR;

    __shared__ float sh[RNK + 2 * NST];   // dts[0..5], B[6..21], C[22..37]

    float dtw[RNK];
    #pragma unroll
    for (int r = 0; r < RNK; ++r) dtw[r] = dt_w[((k * DIN) + d) * RNK + r];
    const float dtb = dt_b[k * DIN + d];
    const float Dd  = Ds[k * DIN + d];
    float Ar[NST];
    #pragma unroll
    for (int n = 0; n < NST; ++n) Ar[n] = -__expf(A_logs[((k * DIN) + d) * NST + n]);

    float h[NST];
    #pragma unroll
    for (int n = 0; n < NST; ++n) h[n] = 0.0f;

    const float* projk = proj + (size_t)k * M_ * 64;

    for (int l = 0; l < L_; ++l) {
        int ll  = (k >= 2) ? (L_ - 1 - l) : l;
        int pos = (k & 1) ? ((ll % H_) * W_ + (ll / H_)) : ll;
        size_t row = (size_t)b * L_ + pos;

        if (d < RNK + 2 * NST) sh[d] = projk[row * 64 + d];
        __syncthreads();

        float dt = dtb;
        #pragma unroll
        for (int r = 0; r < RNK; ++r) dt += sh[r] * dtw[r];
        dt = softplusf(dt);

        float u  = XCF[row * DIN + d];
        float du = dt * u;
        float y  = 0.0f;
        #pragma unroll
        for (int n = 0; n < NST; ++n) {
            float dA = __expf(dt * Ar[n]);
            h[n] = dA * h[n] + du * sh[RNK + n];
            y += h[n] * sh[RNK + NST + n];
        }
        atomicAdd(&Yc[row * DIN + d], y + Dd * u);
        __syncthreads();
    }
}

// ---------------- launch ----------------
extern "C" void kernel_launch(void* const* d_in, const int* in_sizes, int n_in,
                              void* d_out, int out_size, void* d_ws, size_t ws_size,
                              hipStream_t stream) {
    const float* x       = (const float*)d_in[0];
    const float* ln1_w   = (const float*)d_in[1];
    const float* ln1_b   = (const float*)d_in[2];
    const float* W_in    = (const float*)d_in[3];
    const float* conv_w  = (const float*)d_in[4];
    const float* conv_b  = (const float*)d_in[5];
    const float* x_projw = (const float*)d_in[6];
    const float* dt_w    = (const float*)d_in[7];
    const float* dt_b    = (const float*)d_in[8];
    const float* A_logs  = (const float*)d_in[9];
    const float* Ds      = (const float*)d_in[10];
    const float* onw     = (const float*)d_in[11];
    const float* onb     = (const float*)d_in[12];
    const float* W_out   = (const float*)d_in[13];
    const float* ln2_w   = (const float*)d_in[14];
    const float* ln2_b   = (const float*)d_in[15];
    const float* fc1_w   = (const float*)d_in[16];
    const float* fc1_b   = (const float*)d_in[17];
    const float* fc2_w   = (const float*)d_in[18];
    const float* fc2_b   = (const float*)d_in[19];

    char* ws = (char*)d_ws;
    size_t off = 0;
    auto alloc = [&](size_t bytes) -> char* {
        char* p = ws + off;
        off = (off + bytes + 255) & ~(size_t)255;
        return p;
    };

    float* xz    = (float*)alloc((size_t)M_ * 384 * 4);
    bf16*  xln   = (bf16*) alloc((size_t)M_ * 96 * 2);
    float* xcf   = (float*)alloc((size_t)M_ * DIN * 4);
    bf16*  xcb   = (bf16*) alloc((size_t)M_ * DIN * 2);
    float* proj  = (float*)alloc((size_t)4 * M_ * 64 * 4);
    float* ycomb = (float*)alloc((size_t)M_ * DIN * 4);
    bf16*  ybuf  = (bf16*) alloc((size_t)M_ * DIN * 2);
    float* hbuf  = (float*)alloc((size_t)M_ * 96 * 4);
    bf16*  a2    = (bf16*) alloc((size_t)M_ * 96 * 2);
    bf16*  m1    = (bf16*) alloc((size_t)M_ * 384 * 2);
    bf16*  Winb  = (bf16*) alloc((size_t)96 * 384 * 2);     // fragment-ordered
    bf16*  xprjb = (bf16*) alloc((size_t)4 * DIN * 64 * 2); // fragment-ordered
    bf16*  Woutb = (bf16*) alloc((size_t)DIN * 96 * 2);     // fragment-ordered
    bf16*  fc1wb = (bf16*) alloc((size_t)96 * 384 * 2);     // fragment-ordered
    bf16*  fc2wb = (bf16*) alloc((size_t)384 * 96 * 2);     // fragment-ordered

    // 1. weight repacks into WMMA B-fragment order (bf16)
    repack_wB_kernel<<<144, 256, 0, stream>>>(W_in,  Winb,  96, 384);
    repack_wB_kernel<<<72,  256, 0, stream>>>(W_out, Woutb, DIN, 96);
    repack_wB_kernel<<<144, 256, 0, stream>>>(fc1_w, fc1wb, 96, 384);
    repack_wB_kernel<<<144, 256, 0, stream>>>(fc2_w, fc2wb, 384, 96);
    repack_xproj_frag_kernel<<<(4 * DIN * 64 + 255) / 256, 256, 0, stream>>>(x_projw, xprjb);

    // 2. LN1 -> bf16
    ln_rows_kernel<<<M_ / 8, 256, 0, stream>>>(x, ln1_w, ln1_b, xln, M_, 96);

    // 3. xz = LN(x) @ W_in   (25088 x 384 x 96)
    gemm_bf16_kernel<0><<<dim3(M_ / 128, 384 / 32), 256, 0, stream>>>(
        xln, Winb, xz, nullptr, nullptr, nullptr, M_, 384, 96, 384);

    // 4. depthwise conv + silu
    dwconv_silu_kernel<<<(B_ * L_ * DIN + 255) / 256, 256, 0, stream>>>(
        xz, conv_w, conv_b, xcf, xcb);

    // 5. 4 direction projections (25088 x 64 x 192), zero-padded N
    for (int k = 0; k < KDIR; ++k) {
        gemm_bf16_kernel<0><<<dim3(M_ / 128, 64 / 32), 256, 0, stream>>>(
            xcb, xprjb + (size_t)k * DIN * 64, proj + (size_t)k * M_ * 64,
            nullptr, nullptr, nullptr, M_, 64, DIN, 64);
    }

    // 6+7. selective scan with atomic combine
    zero_f32_kernel<<<2048, 256, 0, stream>>>(ycomb, M_ * DIN);
    scan_kernel<<<B_ * KDIR, DIN, 0, stream>>>(proj, xcf, dt_w, dt_b, A_logs, Ds, ycomb);

    // 8. out-norm * silu(z) -> bf16
    outnorm_silu_kernel<<<M_ / 8, 256, 0, stream>>>(ycomb, onw, onb, xz, ybuf);

    // 9. h = x + ybuf @ W_out  (25088 x 96 x 192)
    gemm_bf16_kernel<1><<<dim3(M_ / 128, 96 / 32), 256, 0, stream>>>(
        ybuf, Woutb, hbuf, nullptr, nullptr, x, M_, 96, DIN, 96);

    // 10. LN2 -> bf16
    ln_rows_kernel<<<M_ / 8, 256, 0, stream>>>(hbuf, ln2_w, ln2_b, a2, M_, 96);

    // 11. m1 = gelu(a2 @ fc1_w + fc1_b) -> bf16  (25088 x 384 x 96)
    gemm_bf16_kernel<2><<<dim3(M_ / 128, 384 / 32), 256, 0, stream>>>(
        a2, fc1wb, nullptr, m1, fc1_b, nullptr, M_, 384, 96, 384);

    // 12. out = h + m1 @ fc2_w + fc2_b  (25088 x 96 x 384)
    gemm_bf16_kernel<3><<<dim3(M_ / 128, 96 / 32), 256, 0, stream>>>(
        m1, fc2wb, (float*)d_out, nullptr, fc2_b, hbuf, M_, 96, 384, 96);

    (void)in_sizes; (void)n_in; (void)out_size; (void)ws_size;
}
